// UnmixerModel_79972291051849
// MI455X (gfx1250) — matrix operand order repaired and checked by reading, not compiled, not run on hardware
//
#include <hip/hip_runtime.h>

// ---------------------------------------------------------------------------
// UnmixerModel fused pipeline for MI455X (gfx1250, wave32, WMMA 16x16x32 f16)
//   z:[N,D] f32, prototypes:[K,D] f32, N=131072 D=512 K=256
//   out = concat(z_recon[N,D], weights[N,K], residual[N,D]) fp32
// ---------------------------------------------------------------------------

typedef _Float16 v16h __attribute__((ext_vector_type(16)));
typedef float    v8f  __attribute__((ext_vector_type(8)));

#define N_ROWS_TOTAL 131072
#define DDIM 512
#define KDIM 256
#define TEMPERATURE 10.0f
#define EPS 1e-12f

#define ROWS_PER_BLOCK 64           // 4 waves x 16-row M-strip
#define MAIN_BLOCK 128
#define ZH_STRIDE 520               // 512 + 8 f16 pad (1040B rows: breaks bank alignment)
#define WH_STRIDE 264               // 256 + 8 f16 pad
#define SIMS_STRIDE 260             // 256 + 4 f32 pad
#define ZH_BYTES   (ROWS_PER_BLOCK * ZH_STRIDE * 2)    // 66560
#define SIMS_BYTES (ROWS_PER_BLOCK * SIMS_STRIDE * 4)  // 66560
#define SMEM_BYTES (ZH_BYTES + SIMS_BYTES + ROWS_PER_BLOCK * 4)

__device__ __forceinline__ float wave_sum(float v) {
#pragma unroll
  for (int m = 16; m >= 1; m >>= 1) v += __shfl_xor(v, m, 32);
  return v;
}
__device__ __forceinline__ float wave_max(float v) {
#pragma unroll
  for (int m = 16; m >= 1; m >>= 1) v = fmaxf(v, __shfl_xor(v, m, 32));
  return v;
}

// A fragment (16x32 f16, MxK): lane<16 -> row=lane, K {kb..kb+7, kb+16..kb+23}
//                              lane>=16 -> row=lane-16, K shifted by +8
__device__ __forceinline__ v16h load_a_lds(const _Float16* lds, int row_base,
                                           int stride, int k_base) {
  const int lane = threadIdx.x & 31;
  const int r = lane & 15, hi = lane >> 4;
  const _Float16* p0 = lds + (row_base + r) * stride + k_base + hi * 8;
  union { uint4 q[2]; v16h v; } u;
  u.q[0] = *(const uint4*)(p0);
  u.q[1] = *(const uint4*)(p0 + 16);
  return u.v;
}

// B fragment (32x16 f16, KxN): lane<16 -> col=n_base+lane, K {kb..kb+7, kb+16..kb+23}
//                              lane>=16 -> col=n_base+lane-16, K shifted by +8
// Memory holds B columns contiguously: g[col*stride + k] -> two 16B loads/lane, 32B apart.
__device__ __forceinline__ v16h load_b_glb(const _Float16* g, int n_base,
                                           int stride, int k_base) {
  const int lane = threadIdx.x & 31;
  const int n = n_base + (lane & 15), hi = lane >> 4;
  const _Float16* p0 = g + (size_t)n * stride + k_base + hi * 8;
  union { uint4 q[2]; v16h v; } u;
  u.q[0] = *(const uint4*)(p0);
  u.q[1] = *(const uint4*)(p0 + 16);
  return u.v;
}

// ---------------------------------------------------------------------------
// Kernel 1: normalize prototypes -> f16 row-major p_h[K][D] and f16
// transposed pT_h[D][K] (so both GEMMs get contiguous B-fragment loads).
// ---------------------------------------------------------------------------
__global__ void __launch_bounds__(128)
proto_norm_kernel(const float* __restrict__ proto,
                  _Float16* __restrict__ p_h, _Float16* __restrict__ pT_h) {
  __shared__ float partial[4];
  const int k = blockIdx.x;            // prototype row
  const int t = threadIdx.x;           // 128 threads, 4 f32 each
  const float4 v = *(const float4*)(proto + (size_t)k * DDIM + t * 4);
  float ss = v.x * v.x + v.y * v.y + v.z * v.z + v.w * v.w;
  ss = wave_sum(ss);
  if ((t & 31) == 0) partial[t >> 5] = ss;
  __syncthreads();
  const float total = partial[0] + partial[1] + partial[2] + partial[3];
  const float rn = 1.0f / fmaxf(sqrtf(total), EPS);
  const float e[4] = {v.x * rn, v.y * rn, v.z * rn, v.w * rn};
#pragma unroll
  for (int i = 0; i < 4; ++i) {
    const int d = t * 4 + i;
    const _Float16 h = (_Float16)e[i];
    p_h[(size_t)k * DDIM + d] = h;
    pT_h[(size_t)d * KDIM + k] = h;
  }
}

// ---------------------------------------------------------------------------
// Kernel 2: fused normalize -> GEMM1(WMMA) -> sparsemax -> GEMM2(WMMA) -> out
// ---------------------------------------------------------------------------
__global__ void __launch_bounds__(MAIN_BLOCK)
unmixer_main_kernel(const float* __restrict__ z,
                    const _Float16* __restrict__ p_h,
                    const _Float16* __restrict__ pT_h,
                    float* __restrict__ out_recon,
                    float* __restrict__ out_w,
                    float* __restrict__ out_res) {
  extern __shared__ char smem[];
  _Float16* zh = (_Float16*)smem;                        // [64][ZH_STRIDE]
  _Float16* wh = (_Float16*)smem;                        // reuses zh after GEMM1
  float* sims  = (float*)(smem + ZH_BYTES);              // [64][SIMS_STRIDE]
  float* rnA   = (float*)(smem + ZH_BYTES + SIMS_BYTES); // [64]

  const int wave = threadIdx.x >> 5;
  const int lane = threadIdx.x & 31;
  const int m0 = wave * 16;                              // wave-local row strip
  const size_t row0 = (size_t)blockIdx.x * ROWS_PER_BLOCK;

  // ---- Phase A: L2-normalize 16 z rows per wave, stash f16 tile in LDS ----
#pragma unroll 1
  for (int lr = m0; lr < m0 + 16; ++lr) {
    const float* zr = z + (row0 + lr) * DDIM + lane * 16;
    float4 f0 = *(const float4*)(zr + 0);
    float4 f1 = *(const float4*)(zr + 4);
    float4 f2 = *(const float4*)(zr + 8);
    float4 f3 = *(const float4*)(zr + 12);
    float ss = f0.x*f0.x + f0.y*f0.y + f0.z*f0.z + f0.w*f0.w
             + f1.x*f1.x + f1.y*f1.y + f1.z*f1.z + f1.w*f1.w
             + f2.x*f2.x + f2.y*f2.y + f2.z*f2.z + f2.w*f2.w
             + f3.x*f3.x + f3.y*f3.y + f3.z*f3.z + f3.w*f3.w;
    ss = wave_sum(ss);
    const float rn = 1.0f / fmaxf(sqrtf(ss), EPS);
    if (lane == 0) rnA[lr] = rn;
    _Float16* dst = zh + lr * ZH_STRIDE + lane * 16;
    const float s[16] = {f0.x,f0.y,f0.z,f0.w, f1.x,f1.y,f1.z,f1.w,
                         f2.x,f2.y,f2.z,f2.w, f3.x,f3.y,f3.z,f3.w};
#pragma unroll
    for (int i = 0; i < 16; ++i) dst[i] = (_Float16)(s[i] * rn);
  }
  __syncthreads();

  // ---- Phase B: logits = 10 * z_norm @ p_normT  (WMMA f16 -> f32) ----
#pragma unroll 1
  for (int khalf = 0; khalf < 2; ++khalf) {
    v8f acc[8];
#pragma unroll
    for (int i = 0; i < 8; ++i) acc[i] = {};
#pragma unroll 1
    for (int dt = 0; dt < 16; ++dt) {                    // D chunks of 32
      const v16h a = load_a_lds(zh, m0, ZH_STRIDE, dt * 32);
#pragma unroll
      for (int kt = 0; kt < 8; ++kt) {
        const v16h b = load_b_glb(p_h, khalf * 128 + kt * 16, DDIM, dt * 32);
        acc[kt] = __builtin_amdgcn_wmma_f32_16x16x32_f16(
            false, a, false, b, (short)0, acc[kt], false, false);
      }
    }
    const int rb = m0 + (lane >> 4) * 8, cl = lane & 15;
#pragma unroll
    for (int kt = 0; kt < 8; ++kt)
#pragma unroll
      for (int j = 0; j < 8; ++j)
        sims[(rb + j) * SIMS_STRIDE + khalf * 128 + kt * 16 + cl] =
            acc[kt][j] * TEMPERATURE;
  }
  __syncthreads();

  // ---- Phase C: exact sparsemax per row (bisection + exact support pass) --
#pragma unroll 1
  for (int lr = m0; lr < m0 + 16; ++lr) {
    const float* sr = sims + lr * SIMS_STRIDE + lane * 8;
    float4 a0 = *(const float4*)(sr);
    float4 a1 = *(const float4*)(sr + 4);
    const float v[8] = {a0.x, a0.y, a0.z, a0.w, a1.x, a1.y, a1.z, a1.w};
    float mx = v[0];
#pragma unroll
    for (int i = 1; i < 8; ++i) mx = fmaxf(mx, v[i]);
    mx = wave_max(mx);
    float lo = mx - 1.0f, hi = mx;          // f(lo) >= 0 > f(hi), f monotone
#pragma unroll 1
    for (int it = 0; it < 26; ++it) {
      const float mid = 0.5f * (lo + hi);
      float s = 0.f;
#pragma unroll
      for (int i = 0; i < 8; ++i) s += fmaxf(v[i] - mid, 0.f);
      s = wave_sum(s);
      if (s >= 1.0f) lo = mid; else hi = mid;
    }
    const float thr = 0.5f * (lo + hi);     // support = {v > thr}, exact tau:
    float cnt = 0.f, ssum = 0.f;
#pragma unroll
    for (int i = 0; i < 8; ++i) {
      if (v[i] > thr) { cnt += 1.f; ssum += v[i]; }
    }
    cnt = fmaxf(wave_sum(cnt), 1.0f);
    ssum = wave_sum(ssum);
    const float tau = (ssum - 1.0f) / cnt;
    float* wo = out_w + (row0 + lr) * KDIM + lane * 8;
    _Float16* wl = wh + lr * WH_STRIDE + lane * 8;
#pragma unroll
    for (int i = 0; i < 8; ++i) {
      const float w = fmaxf(v[i] - tau, 0.f);
      wo[i] = w;
      wl[i] = (_Float16)w;                  // overwrites retired zh region
    }
  }
  __syncthreads();

  // ---- Phase D: z_recon = weights @ p_norm; residual = z_norm - z_recon ---
#pragma unroll 1
  for (int ng = 0; ng < 8; ++ng) {                       // 64 output cols each
    v8f acc[4];
#pragma unroll
    for (int i = 0; i < 4; ++i) acc[i] = {};
#pragma unroll 1
    for (int kt = 0; kt < 8; ++kt) {                     // K chunks of 32
      const v16h a = load_a_lds(wh, m0, WH_STRIDE, kt * 32);
#pragma unroll
      for (int nt = 0; nt < 4; ++nt) {
        const v16h b = load_b_glb(pT_h, ng * 64 + nt * 16, KDIM, kt * 32);
        acc[nt] = __builtin_amdgcn_wmma_f32_16x16x32_f16(
            false, a, false, b, (short)0, acc[nt], false, false);
      }
    }
    const int rb = m0 + (lane >> 4) * 8, cl = lane & 15;
#pragma unroll
    for (int nt = 0; nt < 4; ++nt) {
#pragma unroll
      for (int j = 0; j < 8; ++j) {
        const int lr = rb + j;
        const size_t rg = row0 + lr;
        const int col = ng * 64 + nt * 16 + cl;
        const float zr = acc[nt][j];
        const float zn = z[rg * DDIM + col] * rnA[lr];   // exact fp32 z_norm
        out_recon[rg * DDIM + col] = zr;
        out_res[rg * DDIM + col] = zn - zr;
      }
    }
  }
}

// ---------------------------------------------------------------------------
extern "C" void kernel_launch(void* const* d_in, const int* in_sizes, int n_in,
                              void* d_out, int out_size, void* d_ws,
                              size_t ws_size, hipStream_t stream) {
  (void)in_sizes; (void)n_in; (void)out_size; (void)ws_size;
  const float* z = (const float*)d_in[0];
  const float* proto = (const float*)d_in[1];

  _Float16* p_h  = (_Float16*)d_ws;                              // [K][D] f16
  _Float16* pT_h = (_Float16*)((char*)d_ws + (size_t)KDIM * DDIM * 2); // [D][K]

  float* out = (float*)d_out;
  float* out_recon = out;                                        // [N,D]
  float* out_w     = out + (size_t)N_ROWS_TOTAL * DDIM;          // [N,K]
  float* out_res   = out_w + (size_t)N_ROWS_TOTAL * KDIM;        // [N,D]

  proto_norm_kernel<<<KDIM, 128, 0, stream>>>(proto, p_h, pT_h);

  const int grid = N_ROWS_TOTAL / ROWS_PER_BLOCK;                // 2048
  unmixer_main_kernel<<<grid, MAIN_BLOCK, SMEM_BYTES, stream>>>(
      z, p_h, pT_h, out_recon, out_w, out_res);
}
